// DimeNetPlusPlusPotential_71047349010948
// MI455X (gfx1250) — matrix operand order repaired and checked by reading, not compiled, not run on hardware
//
#include <hip/hip_runtime.h>
#include <math.h>

// ---------------- model dimensions ----------------
#define CUTOFF_R 5.0f
#define HD  128   // hidden
#define IED 64    // int_emb
#define OED 256   // out_emb
#define LDm 7     // L
#define KDm 6     // K
#define LKD 42    // L*K
#define GDm 256   // num graphs (B_MOL)

typedef __attribute__((ext_vector_type(16))) _Float16 v16h;
typedef __attribute__((ext_vector_type(8)))  _Float16 v8h;
typedef __attribute__((ext_vector_type(8)))  float    v8f;

struct BesselConsts { float zeros[LDm][KDm]; float norms[LDm][KDm]; float sphn[LDm]; };

// ---------------- device helpers ----------------
__device__ __forceinline__ float dsig(float x) { return 1.f / (1.f + __expf(-x)); }

__device__ __forceinline__ void sphj_pair(float y, int l, float& jl, float& jlm1) {
  float sy = sinf(y), cy = cosf(y);
  float j0 = sy / y;
  if (l == 0) { jl = j0; jlm1 = cy / y; return; }
  float j1 = sy / (y * y) - cy / y;
  float jm = j0, jc = j1;
  for (int n = 1; n < l; ++n) { float t = (2.f * n + 1.f) / y * jc - jm; jm = jc; jc = t; }
  jl = jc; jlm1 = jm;
}
__device__ __forceinline__ float sphj(float y, int l) { float a, b; sphj_pair(y, l, a, b); return a; }

__device__ __forceinline__ void envelope6(float x, float& env, float& denv) {
  const float a = -28.f, b = 48.f, c = -21.f; // p = 6
  float x2 = x * x, x4 = x2 * x2, x5 = x4 * x;
  env  = 1.f / x + a * x5 + b * x5 * x + c * x5 * x2;
  denv = -1.f / x2 + 5.f * a * x4 + 6.f * b * x5 + 7.f * c * x5 * x;
  if (x >= 1.f) { env = 0.f; denv = 0.f; }
}

// ---------------- weight packing: f32 -> f16, zero padded ----------------
__global__ void dn_pack_f16(_Float16* dst, int ldd, long ntot,
                            const float* src, int lds, int R, int C, int trans) {
  long t = (long)blockIdx.x * 256 + threadIdx.x; if (t >= ntot) return;
  long r = t / ldd; int c = (int)(t - r * ldd);
  float v = 0.f;
  if (r < R && c < C) v = trans ? src[(long)c * lds + r] : src[(long)r * lds + c];
  dst[t] = (_Float16)v;
}

// ---------------- fast WMMA GEMM ----------------
// C[M x N] = accC*C + act(A[M x Kd] @ Beff + bias)
// Bt f16 [Nr x Kd] row-major with Bt[n][k] = Beff[k][n]; Nr >= roundup32(N),
// Kd = ldb = roundup32(true K), zero padded. One wave = 16x32 C strip.
__global__ __launch_bounds__(256)
void dn_gemm_fast(const float* __restrict__ A, int lda,
                  const _Float16* __restrict__ Bt, int ldb,
                  const float* __restrict__ bias,
                  float* __restrict__ C, int ldc,
                  float* __restrict__ preOut,
                  int M, int N, int Kd, int act, int accC)
{
  const int lane = threadIdx.x & 31;
  const int wave = threadIdx.x >> 5;
  const int tilesN = (N + 31) >> 5;
  const long numTiles = (long)((M + 15) >> 4) * tilesN;
  const long tile = (long)blockIdx.x * 8 + wave;
  if (tile >= numTiles) return;
  const int tm = (int)(tile / tilesN) << 4;
  const int tn = (int)(tile % tilesN) << 5;
  const int half = lane >> 4;
  const int l15  = lane & 15;

  const float* Arow = A + (long)(tm + l15) * lda + 16 * half;
  const _Float16* B0 = Bt + (long)(tn + l15) * ldb + 16 * half;
  const _Float16* B1 = Bt + (long)(tn + 16 + l15) * ldb + 16 * half;

  v8f acc0 = {}, acc1 = {};
  for (int k0 = 0; k0 < Kd; k0 += 32) {
    if (k0 + 32 < Kd) __builtin_prefetch(Arow + k0 + 32);   // global_prefetch_b8
    v16h af;
#pragma unroll
    for (int t = 0; t < 8; ++t) {
      af[t]     = (_Float16)Arow[k0 + t];
      af[t + 8] = (_Float16)Arow[k0 + 8 + t];
    }
    v8h b0lo = *(const v8h*)(B0 + k0);
    v8h b0hi = *(const v8h*)(B0 + k0 + 8);
    v8h b1lo = *(const v8h*)(B1 + k0);
    v8h b1hi = *(const v8h*)(B1 + k0 + 8);
    v16h bf0 = __builtin_shufflevector(b0lo, b0hi, 0,1,2,3,4,5,6,7,8,9,10,11,12,13,14,15);
    v16h bf1 = __builtin_shufflevector(b1lo, b1hi, 0,1,2,3,4,5,6,7,8,9,10,11,12,13,14,15);
    acc0 = __builtin_amdgcn_wmma_f32_16x16x32_f16(false, af, false, bf0, (short)0, acc0, false, false);
    acc1 = __builtin_amdgcn_wmma_f32_16x16x32_f16(false, af, false, bf1, (short)0, acc1, false, false);
  }

  const int crowBase = tm + 8 * half;
#pragma unroll
  for (int s = 0; s < 2; ++s) {
    v8f av = (s == 0) ? acc0 : acc1;
    int ccol = tn + 16 * s + l15;
    if (ccol >= N) continue;
    float bv = bias ? bias[ccol] : 0.f;
#pragma unroll
    for (int r = 0; r < 8; ++r) {
      int crow = crowBase + r;
      if (crow >= M) continue;
      long off = (long)crow * ldc + ccol;
      float pre = av[r] + bv;
      if (preOut) preOut[off] = pre;
      float val = pre;
      if (act == 1) val = pre * dsig(pre);
      if (accC) val += C[off];
      C[off] = val;
    }
  }
}

// ---------------- small masked WMMA GEMM (tiny shapes only, f32 B) ----------------
__global__ __launch_bounds__(256)
void dn_gemm_small(const float* __restrict__ A, int lda,
                   const float* __restrict__ B, int ldb, int transB,
                   const float* __restrict__ bias,
                   float* __restrict__ C, int ldc,
                   float* __restrict__ preOut,
                   int M, int N, int Kd, int act, int accC)
{
  const int lane = threadIdx.x & 31;
  const int wave = threadIdx.x >> 5;
  const int tilesN = (N + 15) >> 4;
  const long numTiles = (long)((M + 15) >> 4) * tilesN;
  const long tile = (long)blockIdx.x * 8 + wave;
  if (tile >= numTiles) return;
  const int tm = (int)(tile / tilesN) << 4;
  const int tn = (int)(tile % tilesN) << 4;
  const int half = lane >> 4;
  const int l15  = lane & 15;

  const int arow = tm + l15;
  const bool aok = arow < M;
  const float* Arow = A + (long)(aok ? arow : 0) * lda;
  const int bcol = tn + l15;
  const bool bok = bcol < N;

  v8f acc = {};
  for (int k0 = 0; k0 < Kd; k0 += 32) {
    v16h af, bf;
#pragma unroll
    for (int t = 0; t < 16; ++t) {
      int kk = k0 + 16 * half + t;
      float v = (aok && kk < Kd) ? Arow[kk] : 0.f;
      af[t] = (_Float16)v;
    }
#pragma unroll
    for (int t = 0; t < 16; ++t) {
      int kk = k0 + 16 * half + t;
      float v = 0.f;
      if (bok && kk < Kd) v = transB ? B[(long)bcol * ldb + kk] : B[(long)kk * ldb + bcol];
      bf[t] = (_Float16)v;
    }
    acc = __builtin_amdgcn_wmma_f32_16x16x32_f16(false, af, false, bf, (short)0, acc, false, false);
  }
  if (!bok) return;
  const float bv = bias ? bias[bcol] : 0.f;
#pragma unroll
  for (int r = 0; r < 8; ++r) {
    int crow = tm + r + 8 * half;
    if (crow >= M) continue;
    long off = (long)crow * ldc + bcol;
    float pre = acc[r] + bv;
    if (preOut) preOut[off] = pre;
    float val = pre;
    if (act == 1) val = pre * dsig(pre);
    if (accC) val += C[off];
    C[off] = val;
  }
}

// ---------------- elementwise / data-movement kernels ----------------
__global__ void dn_fill(float* p, float v, long n) {
  long t = (long)blockIdx.x * 256 + threadIdx.x; if (t < n) p[t] = v;
}
__global__ void dn_copy(float* d, const float* s, long n) {
  long t = (long)blockIdx.x * 256 + threadIdx.x; if (t < n) d[t] = s[t];
}
__global__ void dn_copyneg(float* d, const float* s, long n) {
  long t = (long)blockIdx.x * 256 + threadIdx.x; if (t < n) d[t] = -s[t];
}
__global__ void dn_add(float* d, const float* s, long n) {
  long t = (long)blockIdx.x * 256 + threadIdx.x; if (t < n) d[t] += s[t];
}
__global__ void dn_mul(float* d, const float* a, const float* b, long n) {
  long t = (long)blockIdx.x * 256 + threadIdx.x; if (t < n) d[t] = a[t] * b[t];
}
__global__ void dn_muladd(float* d, const float* a, const float* b, long n) {
  long t = (long)blockIdx.x * 256 + threadIdx.x; if (t < n) d[t] += a[t] * b[t];
}
__global__ void dn_swishf(float* d, const float* u, long n) {
  long t = (long)blockIdx.x * 256 + threadIdx.x; if (t >= n) return;
  float uu = u[t]; d[t] = uu * dsig(uu);
}
__global__ void dn_swishb(float* d, const float* g, const float* u, long n) {
  long t = (long)blockIdx.x * 256 + threadIdx.x; if (t >= n) return;
  float uu = u[t]; float s = dsig(uu);
  d[t] = g[t] * s * (1.f + uu * (1.f - s));
}
__global__ void dn_gather_rows(float* d, const float* s, const int* idx, long R, int C) {
  long t = (long)blockIdx.x * 256 + threadIdx.x; long n = R * C; if (t >= n) return;
  long r = t / C; int c = (int)(t - r * C);
  d[t] = s[(long)idx[r] * C + c];
}
__global__ void dn_gather_mul(float* d, const float* s, const int* idx, const float* m, long R, int C) {
  long t = (long)blockIdx.x * 256 + threadIdx.x; long n = R * C; if (t >= n) return;
  long r = t / C; int c = (int)(t - r * C);
  d[t] = s[(long)idx[r] * C + c] * m[t];
}
__global__ void dn_scatter_add(float* d, const float* s, const int* idx, long R, int C) {
  long t = (long)blockIdx.x * 256 + threadIdx.x; long n = R * C; if (t >= n) return;
  long r = t / C; int c = (int)(t - r * C);
  atomicAdd(&d[(long)idx[r] * C + c], s[t]);
}
__global__ void dn_scatter_add_mul(float* d, const int* idx, const float* g, const float* m, long R, int C) {
  long t = (long)blockIdx.x * 256 + threadIdx.x; long n = R * C; if (t >= n) return;
  long r = t / C; int c = (int)(t - r * C);
  atomicAdd(&d[(long)idx[r] * C + c], g[t] * m[t]);
}
__global__ void dn_copy_strided(float* d, int dld, const float* s, int sld, long R, int C) {
  long t = (long)blockIdx.x * 256 + threadIdx.x; long n = R * C; if (t >= n) return;
  long r = t / C; int c = (int)(t - r * C);
  d[r * (long)dld + c] = s[r * (long)sld + c];
}
__global__ void dn_concat3(float* cat, const float* xa, const float* hrbf,
                           const int* colI, const int* rowJ, int E) {
  long t = (long)blockIdx.x * 256 + threadIdx.x; long n = (long)E * 3 * HD; if (t >= n) return;
  long e = t / (3 * HD); int c = (int)(t - e * (3 * HD));
  float v;
  if (c < HD)          v = xa[(long)colI[e] * HD + c];
  else if (c < 2 * HD) v = xa[(long)rowJ[e] * HD + (c - HD)];
  else                 v = hrbf[e * (long)HD + (c - 2 * HD)];
  cat[t] = v;
}

// ---------------- geometry: edges ----------------
__global__ void dn_edge_fwd(const float* __restrict__ pos, const int* rowJ, const int* colI,
                            const float* __restrict__ freq, BesselConsts bc,
                            float* __restrict__ rbf, float* __restrict__ rbfs, int E) {
  int e = blockIdx.x * 256 + threadIdx.x; if (e >= E) return;
  int i = colI[e], j = rowJ[e];
  float dx = pos[3*i] - pos[3*j], dy = pos[3*i+1] - pos[3*j+1], dz = pos[3*i+2] - pos[3*j+2];
  float dist = sqrtf(dx*dx + dy*dy + dz*dz);
  float x = dist * (1.f / CUTOFF_R);
  float env, denv; envelope6(x, env, denv);
  for (int k = 0; k < KDm; ++k) rbf[(long)e * KDm + k] = env * sinf(freq[k] * x);
  for (int l = 0; l < LDm; ++l)
    for (int k = 0; k < KDm; ++k) {
      float y = bc.zeros[l][k] * x;
      rbfs[(long)e * LKD + l * KDm + k] = env * bc.norms[l][k] * sphj(y, l);
    }
}

__global__ void dn_edge_bwd(const float* __restrict__ pos, const int* rowJ, const int* colI,
                            const float* __restrict__ freq, BesselConsts bc,
                            const float* __restrict__ d_rbf, const float* __restrict__ d_rbfs,
                            float* __restrict__ dpos, int E) {
  int e = blockIdx.x * 256 + threadIdx.x; if (e >= E) return;
  int i = colI[e], j = rowJ[e];
  float dx = pos[3*i] - pos[3*j], dy = pos[3*i+1] - pos[3*j+1], dz = pos[3*i+2] - pos[3*j+2];
  float dist = sqrtf(dx*dx + dy*dy + dz*dz);
  float x = dist * (1.f / CUTOFF_R);
  float env, denv; envelope6(x, env, denv);
  float gx = 0.f;
  for (int k = 0; k < KDm; ++k) {
    float f = freq[k], s = sinf(f * x), c = cosf(f * x);
    gx += d_rbf[(long)e * KDm + k] * (denv * s + env * f * c);
  }
  for (int l = 0; l < LDm; ++l)
    for (int k = 0; k < KDm; ++k) {
      float z = bc.zeros[l][k], y = z * x;
      float jl, jm; sphj_pair(y, l, jl, jm);
      float jp = jm - (float)(l + 1) / y * jl;
      gx += d_rbfs[(long)e * LKD + l * KDm + k] * bc.norms[l][k] * (denv * jl + env * z * jp);
    }
  float gdist = gx * (1.f / CUTOFF_R);
  float inv = 1.f / dist;
  float ux = dx * inv, uy = dy * inv, uz = dz * inv;
  atomicAdd(&dpos[3*i],   gdist * ux); atomicAdd(&dpos[3*i+1], gdist * uy); atomicAdd(&dpos[3*i+2], gdist * uz);
  atomicAdd(&dpos[3*j],  -gdist * ux); atomicAdd(&dpos[3*j+1],-gdist * uy); atomicAdd(&dpos[3*j+2],-gdist * uz);
}

// ---------------- geometry: triplets ----------------
__global__ void dn_trip_fwd(const float* __restrict__ pos, const int* ti, const int* tj, const int* tk,
                            const int* idxkj, const float* __restrict__ rbfs, BesselConsts bc,
                            float* __restrict__ sbf, int T) {
  int t = blockIdx.x * 256 + threadIdx.x; if (t >= T) return;
  int i = ti[t], j = tj[t], k = tk[t];
  float cx = pos[3*i], cy = pos[3*i+1], cz = pos[3*i+2];
  float jx = pos[3*j]-cx, jy = pos[3*j+1]-cy, jz = pos[3*j+2]-cz;
  float kx = pos[3*k]-cx, ky = pos[3*k+1]-cy, kz = pos[3*k+2]-cz;
  float a = jx*kx + jy*ky + jz*kz;
  float crx = jy*kz - jz*ky, cry = jz*kx - jx*kz, crz = jx*ky - jy*kx;
  float bb = crx*crx + cry*cry + crz*crz;
  float b = sqrtf(fmaxf(bb, 1e-18f));
  float cth = cosf(atan2f(b, a));
  float Pv[LDm]; Pv[0] = 1.f; Pv[1] = cth;
  for (int l = 1; l < LDm - 1; ++l) Pv[l+1] = ((2.f*l+1.f) * cth * Pv[l] - (float)l * Pv[l-1]) / (float)(l+1);
  float cbf[LDm]; cbf[0] = bc.sphn[0];
  for (int l = 1; l < LDm; ++l) cbf[l] = bc.sphn[l] * Pv[l];
  long kj = idxkj[t];
  const float* rs = rbfs + kj * LKD;
  float* sb = sbf + (long)t * LKD;
  for (int l = 0; l < LDm; ++l)
    for (int kk = 0; kk < KDm; ++kk) sb[l * KDm + kk] = rs[l * KDm + kk] * cbf[l];
}

__global__ void dn_trip_bwd(const float* __restrict__ pos, const int* ti, const int* tj, const int* tk,
                            const int* idxkj, const float* __restrict__ rbfs, BesselConsts bc,
                            const float* __restrict__ d_sbf, float* __restrict__ d_rbfs,
                            float* __restrict__ dpos, int T) {
  int t = blockIdx.x * 256 + threadIdx.x; if (t >= T) return;
  int i = ti[t], j = tj[t], k = tk[t];
  float cx = pos[3*i], cy = pos[3*i+1], cz = pos[3*i+2];
  float jx = pos[3*j]-cx, jy = pos[3*j+1]-cy, jz = pos[3*j+2]-cz;
  float kx = pos[3*k]-cx, ky = pos[3*k+1]-cy, kz = pos[3*k+2]-cz;
  float a = jx*kx + jy*ky + jz*kz;
  float crx = jy*kz - jz*ky, cry = jz*kx - jx*kz, crz = jx*ky - jy*kx;
  float bb = crx*crx + cry*cry + crz*crz;
  float b = sqrtf(fmaxf(bb, 1e-18f));
  float ang = atan2f(b, a);
  float cth = cosf(ang);
  float Pv[LDm]; Pv[0] = 1.f; Pv[1] = cth;
  for (int l = 1; l < LDm - 1; ++l) Pv[l+1] = ((2.f*l+1.f) * cth * Pv[l] - (float)l * Pv[l-1]) / (float)(l+1);
  float cbf[LDm]; cbf[0] = bc.sphn[0];
  for (int l = 1; l < LDm; ++l) cbf[l] = bc.sphn[l] * Pv[l];
  long kj = idxkj[t];
  const float* rs = rbfs + kj * LKD;
  const float* ds = d_sbf + (long)t * LKD;
  float dcbf[LDm];
  for (int l = 0; l < LDm; ++l) {
    float acc = 0.f;
    for (int kk = 0; kk < KDm; ++kk) {
      float g = ds[l * KDm + kk];
      acc += g * rs[l * KDm + kk];
      atomicAdd(&d_rbfs[kj * LKD + l * KDm + kk], g * cbf[l]);
    }
    dcbf[l] = acc;
  }
  float Pp[LDm]; Pp[0] = 0.f; Pp[1] = 1.f;
  for (int l = 1; l < LDm - 1; ++l)
    Pp[l+1] = ((2.f*l+1.f) * (Pv[l] + cth * Pp[l]) - (float)l * Pp[l-1]) / (float)(l+1);
  float dcth = 0.f;
  for (int l = 1; l < LDm; ++l) dcth += dcbf[l] * bc.sphn[l] * Pp[l];
  float dang = -sinf(ang) * dcth;
  float denom = a * a + b * b;
  float da = -b / denom * dang;
  float db =  a / denom * dang;
  float gjx = da * kx, gjy = da * ky, gjz = da * kz;
  float gkx = da * jx, gky = da * jy, gkz = da * jz;
  if (bb > 1e-18f) {
    float s = db / b;
    float dcx = s * crx, dcy = s * cry, dcz = s * crz;
    gjx += ky * dcz - kz * dcy; gjy += kz * dcx - kx * dcz; gjz += kx * dcy - ky * dcx;
    gkx += dcy * jz - dcz * jy; gky += dcz * jx - dcx * jz; gkz += dcx * jy - dcy * jx;
  }
  atomicAdd(&dpos[3*j],   gjx); atomicAdd(&dpos[3*j+1], gjy); atomicAdd(&dpos[3*j+2], gjz);
  atomicAdd(&dpos[3*k],   gkx); atomicAdd(&dpos[3*k+1], gky); atomicAdd(&dpos[3*k+2], gkz);
  atomicAdd(&dpos[3*i], -(gjx + gkx)); atomicAdd(&dpos[3*i+1], -(gjy + gky)); atomicAdd(&dpos[3*i+2], -(gjz + gkz));
}

// ---------------- host: spherical Bessel zeros/norms ----------------
static double jn_np_h(double x, int n) {
  double j0 = sin(x) / x;
  if (n == 0) return j0;
  double j1 = sin(x) / (x * x) - cos(x) / x;
  double jm = j0, jc = j1;
  for (int kk = 1; kk < n; ++kk) { double t = (2.0 * kk + 1.0) / x * jc - jm; jm = jc; jc = t; }
  return jc;
}
static void computeBessel(BesselConsts& bc) {
  double zerosj[LDm][KDm];
  for (int k = 0; k < KDm; ++k) zerosj[0][k] = (k + 1) * M_PI;
  double points[KDm + LDm]; int npn = KDm + LDm - 1;
  for (int j = 0; j < npn; ++j) points[j] = (j + 1) * M_PI;
  for (int i = 1; i < LDm; ++i) {
    int nr = npn - 1;
    double rac[KDm + LDm];
    for (int j = 0; j < nr; ++j) {
      double lo = points[j], hi = points[j + 1];
      double flo = jn_np_h(lo, i);
      for (int it = 0; it < 100; ++it) {
        double mid = 0.5 * (lo + hi);
        if (flo * jn_np_h(mid, i) <= 0.0) hi = mid;
        else { lo = mid; flo = jn_np_h(mid, i); }
      }
      rac[j] = 0.5 * (lo + hi);
    }
    for (int j = 0; j < nr; ++j) points[j] = rac[j];
    npn = nr;
    for (int k = 0; k < KDm; ++k) zerosj[i][k] = rac[k];
  }
  for (int l = 0; l < LDm; ++l)
    for (int k = 0; k < KDm; ++k) {
      bc.zeros[l][k] = (float)zerosj[l][k];
      double jv = jn_np_h(zerosj[l][k], l + 1);
      bc.norms[l][k] = (float)(1.0 / sqrt(0.5 * jv * jv));
    }
  for (int l = 0; l < LDm; ++l) bc.sphn[l] = (float)sqrt((2.0 * l + 1.0) / (4.0 * M_PI));
}

// ---------------- host: parameter views ----------------
struct LinP { const float* b; const float* w; };
struct ResP { LinP l1, l2; };
struct IntP {
  ResP after[2]; ResP before[1];
  LinP lin; const float* lin_down;
  LinP lin_ji; LinP lin_kj;
  const float *rbf1, *rbf2, *sbf1, *sbf2, *up;
};
struct OutP { const float* lin; const float* lin_rbf; const float* lin_up; LinP lins[3]; };
struct PackedW { const _Float16* f; int ldf; const _Float16* b; int ldb; };

static inline unsigned eb(long n) { return (unsigned)((n + 255) / 256); }
static inline int ru32i(int x) { return (x + 31) & ~31; }

// ---------------- launch ----------------
extern "C" void kernel_launch(void* const* d_in, const int* in_sizes, int n_in,
                              void* d_out, int out_size, void* d_ws, size_t ws_size,
                              hipStream_t stream) {
  (void)n_in; (void)out_size; (void)ws_size;
  const float* pos  = (const float*)d_in[0];
  const int* zi     = (const int*)d_in[1];
  const int* batch  = (const int*)d_in[2];
  const int* rowJ   = (const int*)d_in[3];
  const int* colI   = (const int*)d_in[4];
  const int* ti     = (const int*)d_in[5];
  const int* tj     = (const int*)d_in[6];
  const int* tk     = (const int*)d_in[7];
  const int* kjI    = (const int*)d_in[8];
  const int* jiI    = (const int*)d_in[9];
  const int N = in_sizes[0] / 3;
  const int E = in_sizes[3];
  const int T = in_sizes[9];
  const int G = GDm;

  int cur = 10;
  auto F = [&]() { return (const float*)d_in[cur++]; };
  const float* emb = F();
  LinP emb_lin; emb_lin.b = F(); emb_lin.w = F();
  LinP emb_rbf; emb_rbf.b = F(); emb_rbf.w = F();
  const float* freq = F();
  LinP head[4];
  for (int h = 0; h < 4; ++h) { head[h].b = F(); head[h].w = F(); }
  IntP ibp[4];
  for (int b = 0; b < 4; ++b) {
    for (int r = 0; r < 2; ++r) {
      ibp[b].after[r].l1.b = F(); ibp[b].after[r].l1.w = F();
      ibp[b].after[r].l2.b = F(); ibp[b].after[r].l2.w = F();
    }
    ibp[b].before[0].l1.b = F(); ibp[b].before[0].l1.w = F();
    ibp[b].before[0].l2.b = F(); ibp[b].before[0].l2.w = F();
    ibp[b].lin.b = F(); ibp[b].lin.w = F();
    ibp[b].lin_down = F();
    ibp[b].lin_ji.b = F(); ibp[b].lin_ji.w = F();
    ibp[b].lin_kj.b = F(); ibp[b].lin_kj.w = F();
    ibp[b].rbf1 = F(); ibp[b].rbf2 = F(); ibp[b].sbf1 = F(); ibp[b].sbf2 = F(); ibp[b].up = F();
  }
  OutP obp[5];
  for (int o = 0; o < 5; ++o) {
    obp[o].lin = F(); obp[o].lin_rbf = F(); obp[o].lin_up = F();
    for (int l = 0; l < 3; ++l) { obp[o].lins[l].b = F(); obp[o].lins[l].w = F(); }
  }
  LinP resL1[4][3], resL2[4][3];
  for (int b = 0; b < 4; ++b) {
    resL1[b][0] = ibp[b].before[0].l1; resL2[b][0] = ibp[b].before[0].l2;
    resL1[b][1] = ibp[b].after[0].l1;  resL2[b][1] = ibp[b].after[0].l2;
    resL1[b][2] = ibp[b].after[1].l1;  resL2[b][2] = ibp[b].after[1].l2;
  }

  BesselConsts bc; computeBessel(bc);

  // ---- workspace ----
  char* wp = (char*)d_ws;
  auto alloc = [&](long nf) -> float* { float* p = (float*)wp; wp += ((nf * 4 + 255) / 256) * 256; return p; };
  auto allocH = [&](long nh) -> _Float16* { return (_Float16*)alloc((nh + 1) / 2); };
  float* rbf      = alloc((long)E * KDm);
  float* rbfs     = alloc((long)E * LKD);
  float* sbf      = alloc((long)T * LKD);
  float* u_embrbf = alloc((long)E * HD);
  float* hrbf     = alloc((long)E * HD);
  float* xaN      = alloc((long)N * HD);
  float* cat      = alloc((long)E * 3 * HD);
  float* u_emb    = alloc((long)E * HD);
  float* xs[5]; for (int s = 0; s < 5; ++s) xs[s] = alloc((long)E * HD);
  float *u_ji[4], *u_kj[4], *rW[4], *u_up[4], *u_lin[4], *u_down[4];
  float *u_resA[4][3], *u_resB[4][3];
  float *R2f[4], *S2f[4];
  for (int b = 0; b < 4; ++b) {
    u_ji[b] = alloc((long)E * HD); u_kj[b] = alloc((long)E * HD); rW[b] = alloc((long)E * HD);
    u_up[b] = alloc((long)E * HD); u_lin[b] = alloc((long)E * HD); u_down[b] = alloc((long)E * IED);
    for (int r = 0; r < 3; ++r) { u_resA[b][r] = alloc((long)E * HD); u_resB[b][r] = alloc((long)E * HD); }
    R2f[b] = alloc((long)32 * HD);
    S2f[b] = alloc((long)64 * IED);
  }
  float* u_out[5][3];
  for (int o = 0; o < 5; ++o) for (int l = 0; l < 3; ++l) u_out[o][l] = alloc((long)N * OED);
  float *u_h0 = alloc((long)G * HD), *u_h1 = alloc((long)G * IED), *u_h2 = alloc((long)G * IED);
  float *P = alloc((long)N * HD), *Pg = alloc((long)G * HD);
  float *hg0 = alloc((long)G * HD), *hg1 = alloc((long)G * IED), *hg2 = alloc((long)G * IED), *yg = alloc(G);
  float *nN = alloc((long)N * HD), *nT = alloc((long)N * OED), *nT2 = alloc((long)N * OED);
  float *tA = alloc((long)E * HD), *tB = alloc((long)E * HD), *tC = alloc((long)E * HD);
  float *tD = alloc((long)E * HD), *tE = alloc((long)E * HD), *tF = alloc((long)E * HD);
  float *e64A = alloc((long)E * IED), *e64B = alloc((long)E * IED), *e64C = alloc((long)E * IED), *agg = alloc((long)E * IED);
  float *sT = alloc((long)T * IED), *yT = alloc((long)T * IED);
  float *dx = alloc((long)E * HD), *dxin = alloc((long)E * HD);
  float *dP = alloc((long)N * HD), *dPg = alloc((long)G * HD);
  float *d_rbf = alloc((long)E * KDm), *d_rbfs = alloc((long)E * LKD), *d_sbfB = alloc((long)T * LKD);
  float *dcat = alloc((long)E * 3 * HD), *dpos = alloc((long)N * 3);
  float *dh0 = alloc((long)G * HD), *dh1 = alloc((long)G * IED), *dh2 = alloc((long)G * IED), *dyg = alloc(G);

  // ---- f16 weight packing ----
  auto makePack = [&](const float* w, int K, int Nn) -> PackedW {
    PackedW p;
    int Kp = ru32i(K), Np = ru32i(Nn);
    _Float16* f = allocH((long)Np * Kp);
    _Float16* b = allocH((long)Kp * Np);
    dn_pack_f16<<<eb((long)Np * Kp), 256, 0, stream>>>(f, Kp, (long)Np * Kp, w, Nn, Nn, K, 1);
    dn_pack_f16<<<eb((long)Kp * Np), 256, 0, stream>>>(b, Np, (long)Kp * Np, w, Nn, K, Nn, 0);
    p.f = f; p.ldf = Kp; p.b = b; p.ldb = Np;
    return p;
  };
  PackedW pw_emb_lin = makePack(emb_lin.w, 3 * HD, HD);
  PackedW pw_emb_rbf = makePack(emb_rbf.w, KDm, HD);
  PackedW pw_head[3];
  pw_head[0] = makePack(head[0].w, HD, HD);
  pw_head[1] = makePack(head[1].w, HD, IED);
  pw_head[2] = makePack(head[2].w, IED, IED);
  PackedW pw_ji[4], pw_kj[4], pw_lin[4], pw_down[4], pw_up[4], pw_res[4][3][2];
  PackedW pw_R2[4], pw_S2[4];
  for (int b = 0; b < 4; ++b) {
    pw_ji[b]   = makePack(ibp[b].lin_ji.w, HD, HD);
    pw_kj[b]   = makePack(ibp[b].lin_kj.w, HD, HD);
    pw_lin[b]  = makePack(ibp[b].lin.w, HD, HD);
    pw_down[b] = makePack(ibp[b].lin_down, HD, IED);
    pw_up[b]   = makePack(ibp[b].up, IED, HD);
    for (int r = 0; r < 3; ++r) {
      pw_res[b][r][0] = makePack(resL1[b][r].w, HD, HD);
      pw_res[b][r][1] = makePack(resL2[b][r].w, HD, HD);
    }
  }
  PackedW pw_olin[5], pw_orbf[5], pw_oup[5], pw_olins[5][3];
  for (int o = 0; o < 5; ++o) {
    pw_olin[o] = makePack(obp[o].lin, OED, HD);
    pw_orbf[o] = makePack(obp[o].lin_rbf, KDm, HD);
    pw_oup[o]  = makePack(obp[o].lin_up, HD, OED);
    for (int l = 0; l < 3; ++l) pw_olins[o][l] = makePack(obp[o].lins[l].w, OED, OED);
  }
  alloc(64);  // tail guard for fast-path A overreads

  // ---- host-side launch helpers ----
  auto gemmF = [&](const float* A, int lda, const PackedW& pw, int fwd, const float* bias,
                   float* C, int ldc, float* pre, long M, int Nn, int act, int acc) {
    const _Float16* Bt = fwd ? pw.f : pw.b;
    int ldbt = fwd ? pw.ldf : pw.ldb;
    long tiles = ((M + 15) / 16) * (long)((Nn + 31) / 32);
    dn_gemm_fast<<<dim3((unsigned)((tiles + 7) / 8)), dim3(256), 0, stream>>>(
        A, lda, Bt, ldbt, bias, C, ldc, pre, (int)M, Nn, ldbt, act, acc);
  };
  auto gemmS = [&](const float* A, int lda, const float* B, int ldb, int tr, const float* bias,
                   float* C, int ldc, float* pre, long M, int Nn, int Kk, int act, int acc) {
    long tiles = ((M + 15) / 16) * (long)((Nn + 15) / 16);
    dn_gemm_small<<<dim3((unsigned)((tiles + 7) / 8)), dim3(256), 0, stream>>>(
        A, lda, B, ldb, tr, bias, C, ldc, pre, (int)M, Nn, Kk, act, acc);
  };
  auto fill   = [&](float* p, float v, long n) { dn_fill<<<eb(n), 256, 0, stream>>>(p, v, n); };
  auto add    = [&](float* d, const float* s, long n) { dn_add<<<eb(n), 256, 0, stream>>>(d, s, n); };
  auto mul    = [&](float* d, const float* a, const float* b, long n) { dn_mul<<<eb(n), 256, 0, stream>>>(d, a, b, n); };
  auto muladd = [&](float* d, const float* a, const float* b, long n) { dn_muladd<<<eb(n), 256, 0, stream>>>(d, a, b, n); };
  auto swishf = [&](float* d, const float* u, long n) { dn_swishf<<<eb(n), 256, 0, stream>>>(d, u, n); };
  auto swishb = [&](float* d, const float* g, const float* u, long n) { dn_swishb<<<eb(n), 256, 0, stream>>>(d, g, u, n); };
  auto gathr  = [&](float* d, const float* s, const int* idx, long R, int C) {
    dn_gather_rows<<<eb(R * (long)C), 256, 0, stream>>>(d, s, idx, R, C); };
  auto scat   = [&](float* d, const float* s, const int* idx, long R, int C) {
    dn_scatter_add<<<eb(R * (long)C), 256, 0, stream>>>(d, s, idx, R, C); };

  auto res_fwd = [&](int b, int r, const float* hIn, float* scr, float* hOut) {
    gemmF(hIn, HD, pw_res[b][r][0], 1, resL1[b][r].b, scr,  HD, u_resA[b][r], E, HD, 1, 0);
    gemmF(scr, HD, pw_res[b][r][1], 1, resL2[b][r].b, hOut, HD, u_resB[b][r], E, HD, 1, 0);
    add(hOut, hIn, (long)E * HD);
  };
  auto res_bwd = [&](int b, int r, float* dH, float* s1, float* s2) {
    swishb(s1, dH, u_resB[b][r], (long)E * HD);
    gemmF(s1, HD, pw_res[b][r][1], 0, nullptr, s2, HD, nullptr, E, HD, 0, 0);
    swishb(s2, s2, u_resA[b][r], (long)E * HD);
    gemmF(s2, HD, pw_res[b][r][0], 0, nullptr, dH, HD, nullptr, E, HD, 0, 1);
  };
  auto out_fwd = [&](int o) {
    const OutP& p = obp[o];
    gemmF(rbf, KDm, pw_orbf[o], 1, nullptr, tB, HD, nullptr, E, HD, 0, 0);         // rW_out
    mul(tA, tB, xs[o], (long)E * HD);
    fill(nN, 0.f, (long)N * HD);
    scat(nN, tA, colI, E, HD);
    gemmF(nN,  HD,  pw_oup[o],      1, nullptr,     nT,  OED, nullptr,     N, OED, 0, 0);
    gemmF(nT,  OED, pw_olins[o][0], 1, p.lins[0].b, nT2, OED, u_out[o][0], N, OED, 1, 0);
    gemmF(nT2, OED, pw_olins[o][1], 1, p.lins[1].b, nT,  OED, u_out[o][1], N, OED, 1, 0);
    gemmF(nT,  OED, pw_olins[o][2], 1, p.lins[2].b, nT2, OED, u_out[o][2], N, OED, 1, 0);
    gemmF(nT2, OED, pw_olin[o],     1, nullptr,     P,   HD,  nullptr,     N, HD,  0, 1); // P +=
  };
  auto out_bwd = [&](int o, float* dxAcc) {
    gemmF(dP, HD, pw_olin[o], 0, nullptr, nT, OED, nullptr, N, OED, 0, 0);         // dT3
    swishb(nT, nT, u_out[o][2], (long)N * OED);
    gemmF(nT,  OED, pw_olins[o][2], 0, nullptr, nT2, OED, nullptr, N, OED, 0, 0);
    swishb(nT2, nT2, u_out[o][1], (long)N * OED);
    gemmF(nT2, OED, pw_olins[o][1], 0, nullptr, nT,  OED, nullptr, N, OED, 0, 0);
    swishb(nT, nT, u_out[o][0], (long)N * OED);
    gemmF(nT,  OED, pw_olins[o][0], 0, nullptr, nT2, OED, nullptr, N, OED, 0, 0);  // dT0
    gemmF(nT2, OED, pw_oup[o],      0, nullptr, nN,  HD,  nullptr, N, HD,  0, 0);  // dTn
    gathr(tA, nN, colI, E, HD);                                                    // dt
    gemmF(rbf, KDm, pw_orbf[o], 1, nullptr, tB, HD, nullptr, E, HD, 0, 0);         // rW_out
    muladd(dxAcc, tA, tB, (long)E * HD);
    mul(tC, tA, xs[o], (long)E * HD);
    gemmF(tC, HD, pw_orbf[o], 0, nullptr, d_rbf, KDm, nullptr, E, KDm, 0, 1);      // d_rbf +=
  };
  auto int_fwd = [&](int b) {
    const IntP& p = ibp[b];
    gemmF(xs[b], HD, pw_ji[b], 1, p.lin_ji.b, tA, HD, u_ji[b], E, HD, 1, 0);
    gemmF(xs[b], HD, pw_kj[b], 1, p.lin_kj.b, tB, HD, u_kj[b], E, HD, 1, 0);
    gemmF(rbf, KDm, pw_R2[b], 1, nullptr, rW[b], HD, nullptr, E, HD, 0, 0);
    mul(tB, tB, rW[b], (long)E * HD);                                              // x_kj2
    gemmF(tB, HD, pw_down[b], 1, nullptr, e64A, IED, u_down[b], E, IED, 1, 0);     // x_kj3
    gemmF(sbf, LKD, pw_S2[b], 1, nullptr, sT, IED, nullptr, T, IED, 0, 0);         // s
    dn_gather_mul<<<eb((long)T * IED), 256, 0, stream>>>(yT, e64A, kjI, sT, T, IED);
    fill(agg, 0.f, (long)E * IED);
    scat(agg, yT, jiI, T, IED);
    gemmF(agg, IED, pw_up[b], 1, nullptr, tC, HD, u_up[b], E, HD, 1, 0);           // x_kj4
    add(tC, tA, (long)E * HD);                                                     // h0
    res_fwd(b, 0, tC, tA, tD);                                                     // h1
    gemmF(tD, HD, pw_lin[b], 1, p.lin.b, tE, HD, u_lin[b], E, HD, 1, 0);
    add(tE, xs[b], (long)E * HD);                                                  // h2
    res_fwd(b, 1, tE, tA, tF);                                                     // h3
    res_fwd(b, 2, tF, tA, xs[b + 1]);                                              // h4 = x_out
  };
  auto int_bwd = [&](int b, float* dxTop, float* dxAcc) {
    res_bwd(b, 2, dxTop, tA, tB);                                                  // dh3
    res_bwd(b, 1, dxTop, tA, tB);                                                  // dh2
    add(dxAcc, dxTop, (long)E * HD);                                               // skip "+ x"
    swishb(tA, dxTop, u_lin[b], (long)E * HD);
    gemmF(tA, HD, pw_lin[b], 0, nullptr, tB, HD, nullptr, E, HD, 0, 0);            // dh1
    res_bwd(b, 0, tB, tA, tC);                                                     // tB = dh0
    swishb(tC, tB, u_up[b], (long)E * HD);
    gemmF(tC, HD, pw_up[b], 0, nullptr, e64A, IED, nullptr, E, IED, 0, 0);         // dagg
    gathr(yT, e64A, jiI, T, IED);                                                  // dy
    gemmF(sbf, LKD, pw_S2[b], 1, nullptr, sT, IED, nullptr, T, IED, 0, 0);         // s (recompute)
    swishf(e64B, u_down[b], (long)E * IED);                                        // x_kj3 (recompute)
    fill(e64C, 0.f, (long)E * IED);
    dn_scatter_add_mul<<<eb((long)T * IED), 256, 0, stream>>>(e64C, kjI, yT, sT, T, IED);
    dn_gather_mul<<<eb((long)T * IED), 256, 0, stream>>>(sT, e64B, kjI, yT, T, IED);
    gemmF(sT, IED, pw_S2[b], 0, nullptr, d_sbfB, LKD, nullptr, T, LKD, 0, 1);      // d_sbf +=
    swishb(e64C, e64C, u_down[b], (long)E * IED);
    gemmF(e64C, IED, pw_down[b], 0, nullptr, tC, HD, nullptr, E, HD, 0, 0);        // d x_kj2
    swishf(tD, u_kj[b], (long)E * HD);                                             // x_kj1
    mul(tE, tC, tD, (long)E * HD);                                                 // d rW
    gemmF(tE, HD, pw_R2[b], 0, nullptr, d_rbf, KDm, nullptr, E, KDm, 0, 1);        // d_rbf +=
    mul(tC, tC, rW[b], (long)E * HD);                                              // d x_kj1
    swishb(tC, tC, u_kj[b], (long)E * HD);
    gemmF(tC, HD, pw_kj[b], 0, nullptr, dxAcc, HD, nullptr, E, HD, 0, 1);
    swishb(tB, tB, u_ji[b], (long)E * HD);
    gemmF(tB, HD, pw_ji[b], 0, nullptr, dxAcc, HD, nullptr, E, HD, 0, 1);
  };

  // ================= FORWARD =================
  dn_edge_fwd<<<eb(E), 256, 0, stream>>>(pos, rowJ, colI, freq, bc, rbf, rbfs, E);
  dn_trip_fwd<<<eb(T), 256, 0, stream>>>(pos, ti, tj, tk, kjI, rbfs, bc, sbf, T);
  for (int b = 0; b < 4; ++b) {
    fill(R2f[b], 0.f, (long)32 * HD);
    fill(S2f[b], 0.f, (long)64 * IED);
    gemmS(ibp[b].rbf1, 8, ibp[b].rbf2, HD,  0, nullptr, R2f[b], HD,  nullptr, KDm, HD,  8, 0, 0);
    gemmS(ibp[b].sbf1, 8, ibp[b].sbf2, IED, 0, nullptr, S2f[b], IED, nullptr, LKD, IED, 8, 0, 0);
    pw_R2[b] = makePack(R2f[b], 32, HD);
    pw_S2[b] = makePack(S2f[b], 64, IED);
  }
  gemmF(rbf, KDm, pw_emb_rbf, 1, emb_rbf.b, hrbf, HD, u_embrbf, E, HD, 1, 0);
  gathr(xaN, emb, zi, N, HD);
  dn_concat3<<<eb((long)E * 3 * HD), 256, 0, stream>>>(cat, xaN, hrbf, colI, rowJ, E);
  gemmF(cat, 3 * HD, pw_emb_lin, 1, emb_lin.b, xs[0], HD, u_emb, E, HD, 1, 0);
  fill(P, 0.f, (long)N * HD);
  out_fwd(0);
  for (int b = 0; b < 4; ++b) { int_fwd(b); out_fwd(b + 1); }
  fill(Pg, 0.f, (long)G * HD);
  scat(Pg, P, batch, N, HD);
  gemmF(Pg,  HD,  pw_head[0], 1, head[0].b, hg0, HD,  u_h0, G, HD,  1, 0);
  gemmF(hg0, HD,  pw_head[1], 1, head[1].b, hg1, IED, u_h1, G, IED, 1, 0);
  gemmF(hg1, IED, pw_head[2], 1, head[2].b, hg2, IED, u_h2, G, IED, 1, 0);
  gemmS(hg2, IED, head[3].w, 1, 0, head[3].b, yg, 1, nullptr, G, 1, IED, 0, 0);
  dn_copy<<<eb(G), 256, 0, stream>>>((float*)d_out, yg, G);

  // ================= BACKWARD (forces) =================
  fill(dpos,   0.f, (long)N * 3);
  fill(d_rbf,  0.f, (long)E * KDm);
  fill(d_rbfs, 0.f, (long)E * LKD);
  fill(d_sbfB, 0.f, (long)T * LKD);
  fill(dyg, 1.f, G);
  gemmS(dyg, 1, head[3].w, 1, 1, nullptr, dh2, IED, nullptr, G, IED, 1, 0, 0);
  swishb(dh2, dh2, u_h2, (long)G * IED);
  gemmF(dh2, IED, pw_head[2], 0, nullptr, dh1, IED, nullptr, G, IED, 0, 0);
  swishb(dh1, dh1, u_h1, (long)G * IED);
  gemmF(dh1, IED, pw_head[1], 0, nullptr, dh0, HD,  nullptr, G, HD,  0, 0);
  swishb(dh0, dh0, u_h0, (long)G * HD);
  gemmF(dh0, HD,  pw_head[0], 0, nullptr, dPg, HD,  nullptr, G, HD,  0, 0);
  gathr(dP, dPg, batch, N, HD);

  fill(dx, 0.f, (long)E * HD);
  out_bwd(4, dx);
  for (int b = 3; b >= 0; --b) {
    fill(dxin, 0.f, (long)E * HD);
    int_bwd(b, dx, dxin);
    out_bwd(b, dxin);
    float* tsw = dx; dx = dxin; dxin = tsw;
  }
  swishb(dx, dx, u_emb, (long)E * HD);
  gemmF(dx, HD, pw_emb_lin, 0, nullptr, dcat, 3 * HD, nullptr, E, 3 * HD, 0, 0);
  dn_copy_strided<<<eb((long)E * HD), 256, 0, stream>>>(tA, HD, dcat + 2 * HD, 3 * HD, E, HD);
  swishb(tA, tA, u_embrbf, (long)E * HD);
  gemmF(tA, HD, pw_emb_rbf, 0, nullptr, d_rbf, KDm, nullptr, E, KDm, 0, 1);

  dn_trip_bwd<<<eb(T), 256, 0, stream>>>(pos, ti, tj, tk, kjI, rbfs, bc, d_sbfB, d_rbfs, dpos, T);
  dn_edge_bwd<<<eb(E), 256, 0, stream>>>(pos, rowJ, colI, freq, bc, d_rbf, d_rbfs, dpos, E);
  dn_copyneg<<<eb((long)N * 3), 256, 0, stream>>>((float*)d_out + G, dpos, (long)N * 3);
}
